// DeepAndWide_79920751444429
// MI455X (gfx1250) — matrix operand order repaired and checked by reading, not compiled, not run on hardware
//
#include <hip/hip_runtime.h>

typedef __attribute__((ext_vector_type(16))) _Float16 v16h;
typedef __attribute__((ext_vector_type(8)))  _Float16 v8h;
typedef __attribute__((ext_vector_type(8)))  float    v8f;
typedef __attribute__((ext_vector_type(4)))  float    v4f;

#define BATCH 1024
#define DDIM  384
#define TWOD  768
#define KKW   (DDIM * DDIM)       // 147456
#define OSTR  1536                // output row stride (4*D)

// ---- wide-path tiling ----
#define MBLK   128                // 8 waves x 16 rows
#define NBLK   128                // 8 WMMA tiles of 16 cols per wave
#define NTILES (NBLK / 16)
#define KSPLIT 8
#define NCHUNK (KKW / 32)         // 4608 K-chunks of 32
#define CPB    (NCHUNK / KSPLIT)  // 576 chunks per block
#define BSTR   40                 // padded LDS B row stride (halves)
#define BBUF   (NBLK * BSTR)      // halves per LDS buffer

// out[b, 768+n] = b_L2[n]  (wide path bias, atomics accumulate on top)
__global__ __launch_bounds__(256) void wide_bias_init(
    const float* __restrict__ bL2, float* __restrict__ out) {
  int idx = blockIdx.x * 256 + threadIdx.x;
  if (idx < BATCH * TWOD) {
    int b = idx / TWOD;
    int n = idx - b * TWOD;
    out[(size_t)b * OSTR + TWOD + n] = bL2[n];
  }
}

// Wide path: u = (hs outer ht) @ W_L2^T, split-K with f32 atomics.
// - W_L2 slab staged to double-buffered LDS, f32->f16 converted once/block
// - B fragments pipelined depth-3 from LDS (loads 2 WMMAs ahead)
// - A source prefetched one chunk ahead; a_next built after WMMAs issue
__global__ __launch_bounds__(256) void wide_kernel(
    const float* __restrict__ hs, const float* __restrict__ ht,
    const float* __restrict__ W2, float* __restrict__ out) {
  __shared__ alignas(16) _Float16 sB[2 * BBUF];  // 2 x (128 x 32) f16, padded

  const int tid  = threadIdx.x;
  const int lane = tid & 31;
  const int wave = tid >> 5;
  const int half = lane >> 4;   // 0: lanes 0-15, 1: lanes 16-31
  const int mrow = lane & 15;

  const int b0  = blockIdx.x * MBLK;
  const int n0  = blockIdx.y * NBLK;
  const int kc0 = blockIdx.z * CPB;

  const int brow = b0 + wave * 16 + mrow;  // A-matrix row of this lane
  const float* hsrow = hs + (size_t)brow * DDIM;
  const float* htrow = ht + (size_t)brow * DDIM;

  // cooperative staging: thread -> (row 0..127, half-of-32 k's)
  const int sr = tid >> 1;
  const int sq = tid & 1;
  const float* wsrc0 = W2 + (size_t)(n0 + sr) * KKW + sq * 16;
  const int sdoff = sr * BSTR + sq * 16;

  v8f acc[NTILES] = {};

  // ---- A-source prefetch / build ----
  struct ARegs { v4f h00, h01, h10, h11; float hsv; };
  auto load_a = [&](int kc) {
    const int i  = kc / 12;            // 384/32 == 12 chunks per i
    const int jb = (kc - i * 12) * 32;
    const float* hj = htrow + jb + half * 16;
    ARegs r;
    r.h00 = *(const v4f*)(hj);
    r.h01 = *(const v4f*)(hj + 4);
    r.h10 = *(const v4f*)(hj + 8);
    r.h11 = *(const v4f*)(hj + 12);
    r.hsv = hsrow[i];
    return r;
  };
  // v[b, i*D + j] = hs[b,i] * ht[b,j]
  // A k order used here (B matches): lane half h, element e -> K = h*16 + e
  auto build_a = [&](const ARegs& r) {
    v16h a;
#pragma unroll
    for (int e = 0; e < 4; ++e) {
      a[e]      = (_Float16)(r.hsv * r.h00[e]);
      a[e + 4]  = (_Float16)(r.hsv * r.h01[e]);
      a[e + 8]  = (_Float16)(r.hsv * r.h10[e]);
      a[e + 12] = (_Float16)(r.hsv * r.h11[e]);
    }
    return a;
  };

  // ---- depth-3 pipelined B fragments + 8 WMMAs ----
  auto mma_tiles = [&](const _Float16* __restrict__ sBr, const v16h& a) {
    const int bb = mrow * BSTR + half * 16;
    v8h q0[3], q1[3];
    q0[0] = *(const v8h*)&sBr[bb];
    q1[0] = *(const v8h*)&sBr[bb + 8];
    q0[1] = *(const v8h*)&sBr[bb + 16 * BSTR];
    q1[1] = *(const v8h*)&sBr[bb + 16 * BSTR + 8];
#pragma unroll
    for (int nt = 0; nt < NTILES; ++nt) {
      const int cur = nt % 3;
      if (nt + 2 < NTILES) {
        const int b2 = bb + (nt + 2) * 16 * BSTR;
        q0[(nt + 2) % 3] = *(const v8h*)&sBr[b2];
        q1[(nt + 2) % 3] = *(const v8h*)&sBr[b2 + 8];
      }
      v16h bf;
#pragma unroll
      for (int e = 0; e < 8; ++e) { bf[e] = q0[cur][e]; bf[e + 8] = q1[cur][e]; }
      acc[nt] = __builtin_amdgcn_wmma_f32_16x16x32_f16(
          false, a, false, bf, (short)0, acc[nt], false, false);
    }
  };

  // ---- prologue: stage first chunk into buffer 0; build first A ----
  ARegs ar0 = load_a(kc0);
  {
    const float* src = wsrc0 + kc0 * 32;
    v4f w0 = *(const v4f*)(src);
    v4f w1 = *(const v4f*)(src + 4);
    v4f w2 = *(const v4f*)(src + 8);
    v4f w3 = *(const v4f*)(src + 12);
    v8h o0, o1;
#pragma unroll
    for (int e = 0; e < 4; ++e) {
      o0[e] = (_Float16)w0[e]; o0[e + 4] = (_Float16)w1[e];
      o1[e] = (_Float16)w2[e]; o1[e + 4] = (_Float16)w3[e];
    }
    *(v8h*)&sB[sdoff] = o0;
    *(v8h*)&sB[sdoff + 8] = o1;
  }
  v16h a_cur = build_a(ar0);
  __syncthreads();

  // ---- main loop ----
  for (int kc = kc0; kc < kc0 + CPB - 1; ++kc) {
    const int pr = (kc - kc0) & 1;
    // prefetch next W slab (global) and next A source (global)
    const float* src = wsrc0 + (kc + 1) * 32;
    v4f w0 = *(const v4f*)(src);
    v4f w1 = *(const v4f*)(src + 4);
    v4f w2 = *(const v4f*)(src + 8);
    v4f w3 = *(const v4f*)(src + 12);
    ARegs an = load_a(kc + 1);

    // compute current chunk
    mma_tiles(sB + pr * BBUF, a_cur);

    // build next A while WMMAs drain
    v16h a_next = build_a(an);

    // convert + store next W slab into the other buffer
    v8h o0, o1;
#pragma unroll
    for (int e = 0; e < 4; ++e) {
      o0[e] = (_Float16)w0[e]; o0[e + 4] = (_Float16)w1[e];
      o1[e] = (_Float16)w2[e]; o1[e + 4] = (_Float16)w3[e];
    }
    _Float16* d = sB + (pr ^ 1) * BBUF + sdoff;
    *(v8h*)d = o0;
    *(v8h*)(d + 8) = o1;
    __syncthreads();  // write-buffer ready AND read-buffer free for next iter
    a_cur = a_next;
  }
  // ---- epilogue: last chunk ----
  mma_tiles(sB + ((CPB - 1) & 1) * BBUF, a_cur);

  // ---- accumulate partial tile: C layout M = r + half*8, N = mrow ----
#pragma unroll
  for (int nt = 0; nt < NTILES; ++nt) {
    const int n = n0 + nt * 16 + mrow;
#pragma unroll
    for (int r = 0; r < 8; ++r) {
      const int m = b0 + wave * 16 + r + half * 8;
      atomicAdd(&out[(size_t)m * OSTR + TWOD + n], acc[nt][r]);
    }
  }
}

// Deep path: g = concat(hs, ht) @ W_L^T + b_L  -> out[:, 0:768]
// 768 wave-tasks: waveid -> (M-tile 0..63, N-block-of-64 0..11)
__global__ __launch_bounds__(256) void deep_kernel(
    const float* __restrict__ hs, const float* __restrict__ ht,
    const float* __restrict__ WL, const float* __restrict__ bL,
    float* __restrict__ out) {
  const int tid  = threadIdx.x;
  const int lane = tid & 31;
  const int wave = tid >> 5;
  const int half = lane >> 4;
  const int mrow = lane & 15;

  const int w  = blockIdx.x * 8 + wave;  // 0..767
  const int mt = w / 12;
  const int nb = w - mt * 12;
  const int brow = mt * 16 + mrow;

  v8f acc[4] = {};

  for (int kc = 0; kc < 24; ++kc) {
    const int kk = kc * 32;  // chunk never crosses the 384 concat boundary
    const float* xr = (kk < DDIM) ? (hs + (size_t)brow * DDIM + kk)
                                  : (ht + (size_t)brow * DDIM + (kk - DDIM));
    const float* ap = xr + half * 16;
    v4f c00 = *(const v4f*)(ap);
    v4f c01 = *(const v4f*)(ap + 4);
    v4f c10 = *(const v4f*)(ap + 8);
    v4f c11 = *(const v4f*)(ap + 12);
    v16h a;
#pragma unroll
    for (int e = 0; e < 4; ++e) {
      a[e]      = (_Float16)c00[e];
      a[e + 4]  = (_Float16)c01[e];
      a[e + 8]  = (_Float16)c10[e];
      a[e + 12] = (_Float16)c11[e];
    }
#pragma unroll
    for (int nt = 0; nt < 4; ++nt) {
      const int n = nb * 64 + nt * 16 + mrow;
      const float* bp = WL + (size_t)n * TWOD + kk + half * 16;
      v4f b0 = *(const v4f*)(bp);
      v4f b1 = *(const v4f*)(bp + 4);
      v4f b2 = *(const v4f*)(bp + 8);
      v4f b3 = *(const v4f*)(bp + 12);
      v16h bf;
#pragma unroll
      for (int e = 0; e < 4; ++e) {
        bf[e]      = (_Float16)b0[e];
        bf[e + 4]  = (_Float16)b1[e];
        bf[e + 8]  = (_Float16)b2[e];
        bf[e + 12] = (_Float16)b3[e];
      }
      acc[nt] = __builtin_amdgcn_wmma_f32_16x16x32_f16(
          false, a, false, bf, (short)0, acc[nt], false, false);
    }
  }

#pragma unroll
  for (int nt = 0; nt < 4; ++nt) {
    const int n = nb * 64 + nt * 16 + mrow;
    const float bias = bL[n];
#pragma unroll
    for (int r = 0; r < 8; ++r) {
      const int m = mt * 16 + r + half * 8;
      out[(size_t)m * OSTR + n] = acc[nt][r] + bias;
    }
  }
}

extern "C" void kernel_launch(void* const* d_in, const int* in_sizes, int n_in,
                              void* d_out, int out_size, void* d_ws, size_t ws_size,
                              hipStream_t stream) {
  (void)in_sizes; (void)n_in; (void)d_ws; (void)ws_size; (void)out_size;
  const float* hs  = (const float*)d_in[0];
  const float* ht  = (const float*)d_in[1];
  const float* WL  = (const float*)d_in[2];
  const float* bL  = (const float*)d_in[3];
  const float* W2  = (const float*)d_in[4];
  const float* bL2 = (const float*)d_in[5];
  float* out = (float*)d_out;

  // 1) wide bias init (must precede wide atomics; same stream => ordered)
  wide_bias_init<<<(BATCH * TWOD + 255) / 256, 256, 0, stream>>>(bL2, out);
  // 2) deep path (independent region of out)
  deep_kernel<<<96, 256, 0, stream>>>(hs, ht, WL, bL, out);
  // 3) wide path main GEMM, split-K over grid.z
  wide_kernel<<<dim3(BATCH / MBLK, TWOD / NBLK, KSPLIT), 256, 0, stream>>>(
      hs, ht, W2, out);
}
